// RoiPooling_88364657148525
// MI455X (gfx1250) — compile-verified, matrix operand order checked by reading, not run on hardware
//
#include <hip/hip_runtime.h>
#include <hip/hip_bf16.h>

typedef __attribute__((ext_vector_type(16))) _Float16 v16h;
typedef __attribute__((ext_vector_type(8)))  _Float16 v8h;
typedef __attribute__((ext_vector_type(8)))  float    v8f;

#define CC 256
#define HH 96
#define WW 96
#define HW (HH * WW)
#define RR 256
#define PE 2
#define PN 4
#define NN (PN * RR)          // 1024 patches
#define KK 20                 // classes
#define PAD 128               // f16 tail padding for aligned overreads
// output section offsets (floats), reference return order
#define OFF_POOL   0
#define OFF_FG     (NN * CC)
#define OFF_NFG    (2 * NN * CC)
#define OFF_LAB    (3 * NN * CC)
#define OFF_SCORE  (3 * NN * CC + NN)
#define OFF_LOCS   (3 * NN * CC + 2 * NN)

// ---------------------------------------------------------------------------
// Kernel 0: stage feature map as f16 and expand cam_predict into 20 f16
// 0/1 class planes. Pads tails with zeros so hot-loop b128 overreads are safe.
// ---------------------------------------------------------------------------
__global__ void rp_stage_f16(const float* __restrict__ F,
                             const int* __restrict__ pred,
                             _Float16* __restrict__ Fh,
                             _Float16* __restrict__ Mh) {
  int idx = blockIdx.x * blockDim.x + threadIdx.x;
  if (idx < CC * HW)            Fh[idx] = (_Float16)F[idx];
  else if (idx < CC * HW + PAD) Fh[idx] = (_Float16)0.0f;
  if (idx < KK * HW) {
    int k = idx / HW, p = idx % HW;
    Mh[idx] = (pred[p] == k) ? (_Float16)1.0f : (_Float16)0.0f;
  } else if (idx < KK * HW + PAD) {
    Mh[idx] = (_Float16)0.0f;
  }
}

// ---------------------------------------------------------------------------
// Kernel 1: per-patch boxes, labels, area, fg pixel count; writes labels/locs.
// ---------------------------------------------------------------------------
__global__ void rp_setup(const float* __restrict__ roi,
                         const int*   __restrict__ label_list,
                         const int*   __restrict__ pred,
                         int* __restrict__ boxes, int* __restrict__ labl,
                         float* __restrict__ cnt, float* __restrict__ area,
                         float* __restrict__ out) {
  int n = blockIdx.x * blockDim.x + threadIdx.x;
  if (n >= NN) return;
  int i = n / RR, r = n % RR;
  float xmin = roi[r * 4 + 0], ymin = roi[r * 4 + 1];
  float xmax = roi[r * 4 + 2], ymax = roi[r * 4 + 3];
  float colf = (float)(i % PE), rowf = (float)(i / PE);
  float wstep = (xmax - xmin) / (float)PE;
  float hstep = (ymax - ymin) / (float)PE;
  float x0f = xmin + colf * wstep;
  float y0f = ymin + rowf * hstep;
  int x0 = (int)floorf(x0f);
  int y0 = (int)floorf(y0f);
  int x1 = (int)ceilf(x0f + wstep); x1 = min(max(x1, 0), WW);
  int y1 = (int)ceilf(y0f + hstep); y1 = min(max(y1, 0), HH);
  int lab = label_list[r];
  int c = 0;
  for (int y = y0; y < y1; ++y)
    for (int x = x0; x < x1; ++x)
      c += (pred[y * WW + x] == lab) ? 1 : 0;
  boxes[4 * n + 0] = x0; boxes[4 * n + 1] = y0;
  boxes[4 * n + 2] = x1; boxes[4 * n + 3] = y1;
  labl[n] = lab;
  cnt[n]  = (float)c;
  area[n] = (float)((y1 - y0) * (x1 - x0));
  out[OFF_LAB + n] = (float)lab;
  out[OFF_LOCS + 4 * n + 0] = (float)x0;
  out[OFF_LOCS + 4 * n + 1] = (float)y0;
  out[OFF_LOCS + 4 * n + 2] = (float)x1;
  out[OFF_LOCS + 4 * n + 3] = (float)y1;
}

// ---------------------------------------------------------------------------
// Kernel 2: WMMA masked reduction. One wave32 per (ROI, 16-channel tile).
// A rows 2i   = membership of patch i (0/1, exact in f16)
// A rows 2i+1 = membership * classmask  (classmask preloaded as f16 plane)
// B           = 32 pixels x 16 channels of f16 features, f32 accumulate.
// Row-wise iteration over the ROI union box, x-grid aligned to 8 so every
// hot-loop load is an aligned b128. Out-of-patch B values are killed by A=0.
// ---------------------------------------------------------------------------
__global__ __launch_bounds__(32)
void rp_wmma_pool(const _Float16* __restrict__ Fh,
                  const _Float16* __restrict__ Mh,
                  const int* __restrict__ boxes, const int* __restrict__ labl,
                  float* __restrict__ out) {
  const int r     = blockIdx.y;          // ROI
  const int cbase = blockIdx.x * 16;     // channel tile
  const int lane  = threadIdx.x;         // 0..31, full wave -> EXEC all ones

  // the 4 patch boxes of this ROI and their union (uniform across the wave)
  int bx0[4], by0[4], bx1[4], by1[4];
  int X0 = WW, Y0 = HH, X1 = 0, Y1 = 0;
#pragma unroll
  for (int i = 0; i < 4; ++i) {
    int n = i * RR + r;
    bx0[i] = boxes[4 * n + 0]; by0[i] = boxes[4 * n + 1];
    bx1[i] = boxes[4 * n + 2]; by1[i] = boxes[4 * n + 3];
    X0 = min(X0, bx0[i]); Y0 = min(Y0, by0[i]);
    X1 = max(X1, bx1[i]); Y1 = max(Y1, by1[i]);
  }
  const int lab = labl[r];               // label shared by all 4 patches
  const int Xa  = X0 & ~7;               // align x-grid: all loads 16B aligned
  const int bwp = X1 - Xa;               // padded row width (pad pixels -> A=0)

  const int m     = lane & 15;           // A row / B column within tile
  const int kb    = (lane >> 4) * 8;     // ISA 16-bit A/B K packing per lane
  const int ch    = cbase + m;           // channel this lane feeds into B
  const bool arow = (m < 8);             // only A rows 0..7 carry data
  const int  pi   = (m >> 1) & 3;        // patch index for this A row
  const bool wfg  = (m & 1) != 0;        // odd rows: foreground-masked
  const int px0 = bx0[pi], px1 = bx1[pi], py0 = by0[pi], py1 = by1[pi];

  const int fpl = ch * HW;               // feature plane base (per lane)
  const int mpl = lab * HW;              // class-mask plane base (uniform)

  v8f acc = {};
  for (int y = Y0; y < Y1; ++y) {
    const bool ay   = arow & (y >= py0) & (y < py1);
    const int  rowf = fpl + y * WW;
    const int  rowm = mpl + y * WW;
    __builtin_prefetch((const void*)(Fh + rowf + WW + Xa), 0, 1);
    for (int xoff = 0; xoff < bwp; xoff += 32) {
      const int xs0 = Xa + xoff + kb;    // multiple of 8 -> aligned b128
      // B: 16 f16 features, two aligned 16B loads (no masking needed)
      v8h b0 = *(const v8h*)(Fh + rowf + xs0);
      v8h b1 = *(const v8h*)(Fh + rowf + xs0 + 16);
      v16h b = __builtin_shufflevector(b0, b1,
                 0, 1, 2, 3, 4, 5, 6, 7, 8, 9, 10, 11, 12, 13, 14, 15);
      // class-mask values for this row segment (same addressing as B)
      v8h m0 = *(const v8h*)(Mh + rowm + xs0);
      v8h m1 = *(const v8h*)(Mh + rowm + xs0 + 16);
      v16h a;
#pragma unroll
      for (int e = 0; e < 16; ++e) {
        int x = xs0 + ((e < 8) ? e : 8 + e);       // k-map: e<8 -> +e, else +16+(e-8)
        bool inx = ay & (x >= px0) & (x < px1);
        _Float16 mv = (e < 8) ? m0[e] : m1[e - 8];
        _Float16 one = (_Float16)1.0f;
        a[e] = inx ? (wfg ? mv : one) : (_Float16)0.0f;
      }
      acc = __builtin_amdgcn_wmma_f32_16x16x32_f16(
          /*neg_a=*/false, a, /*neg_b=*/false, b,
          /*c_mod=*/(short)0, acc, /*reuse_a=*/false, /*reuse_b=*/false);
    }
  }

  // D layout: lane<16 holds rows 0..7 in acc[0..7], column = lane
  if (lane < 16) {
    int c = cbase + lane;
#pragma unroll
    for (int v = 0; v < 8; ++v) {
      int n = (v >> 1) * RR + r;
      if ((v & 1) == 0) out[OFF_POOL + n * CC + c] = acc[v];  // raw feat_sum
      else              out[OFF_FG   + n * CC + c] = acc[v];  // raw fg_sum
    }
  }
}

// ---------------------------------------------------------------------------
// Kernel 3: finalize pools in place + fg_score.
// ---------------------------------------------------------------------------
__global__ void rp_finalize(const float* __restrict__ cnt,
                            const float* __restrict__ area,
                            float* __restrict__ out) {
  int idx = blockIdx.x * blockDim.x + threadIdx.x;
  if (idx >= NN * CC) return;
  int n = idx / CC, c = idx % CC;
  float fs = out[OFF_POOL + idx];
  float fg = out[OFF_FG + idx];
  float ct = cnt[n];
  float ar = area[n];
  out[OFF_POOL + idx] = fs / ar;
  out[OFF_FG + idx]   = (ct > 0.0f) ? fg / fmaxf(ct, 1.0f) : 0.0f;
  float nc = ar - ct;
  out[OFF_NFG + idx]  = (nc > 0.0f) ? (fs - fg) / fmaxf(nc, 1.0f) : 0.0f;
  if (c == 0) out[OFF_SCORE + n] = ct / ar;
}

// ---------------------------------------------------------------------------
extern "C" void kernel_launch(void* const* d_in, const int* in_sizes, int n_in,
                              void* d_out, int out_size, void* d_ws, size_t ws_size,
                              hipStream_t stream) {
  const float* feature_map = (const float*)d_in[0];   // (256,96,96)
  const float* roi_batch   = (const float*)d_in[1];   // (256,4)
  const int*   label_list  = (const int*)d_in[2];     // (256,)
  // d_in[3] = patch_nums (scalar, ==4); d_in[5]/d_in[6] (cam, cls_label) dead
  const int*   cam_predict = (const int*)d_in[4];     // (96,96)
  float* out = (float*)d_out;

  // workspace partition (~5.1 MB)
  _Float16* Fh = (_Float16*)d_ws;                 // CC*HW + PAD halves
  _Float16* Mh = Fh + (CC * HW + PAD);            // KK*HW + PAD halves
  float* cnt  = (float*)(Mh + (KK * HW + PAD));
  float* area = cnt + NN;
  int*   boxes = (int*)(area + NN);
  int*   labl  = boxes + 4 * NN;

  int stage_n = CC * HW + PAD;
  rp_stage_f16<<<(stage_n + 255) / 256, 256, 0, stream>>>(
      feature_map, cam_predict, Fh, Mh);

  rp_setup<<<(NN + 255) / 256, 256, 0, stream>>>(
      roi_batch, label_list, cam_predict, boxes, labl, cnt, area, out);

  dim3 grid2(CC / 16, RR);   // 16 channel tiles x 256 ROIs, one wave32 each
  rp_wmma_pool<<<grid2, 32, 0, stream>>>(Fh, Mh, boxes, labl, out);

  rp_finalize<<<(NN * CC + 255) / 256, 256, 0, stream>>>(cnt, area, out);
}
